// GraphMultiHeadAttention_66640712564942
// MI455X (gfx1250) — matrix operand order repaired and checked by reading, not compiled, not run on hardware
//
#include <hip/hip_runtime.h>
#include <hip/hip_bf16.h>

// ---------------- problem constants ----------------
#define Bdim     4
#define Ndim     1024
#define INFEA    512
#define HIDDEN   128
#define NHEAD    4
#define NEDGE    5
#define Mrows    (Bdim * Ndim)          // 4096
#define NEG_INF  (-1.0e9f)
#define LN_EPS   1e-6f
#define SLOPE    0.01f

typedef __attribute__((ext_vector_type(16))) __bf16 v16bf;
typedef __attribute__((ext_vector_type(8)))  float  v8f;

union Frag16 { v16bf v; unsigned int u[8]; };

// ---------------- Tensor Data Mover helpers ----------------
typedef __attribute__((ext_vector_type(4))) unsigned int tdm4u;
typedef __attribute__((ext_vector_type(8))) int          tdm8i;
typedef __attribute__((ext_vector_type(4))) int          tdm4i;

// D# group 0: count=1 | lds_addr | 57-bit global addr | type=2 ("image")
__device__ __forceinline__ tdm4u tdm_make_g0(unsigned lds_addr, const void* gptr)
{
    unsigned long long ga = (unsigned long long)(uintptr_t)gptr;
    tdm4u g0;
    g0.x = 1u;                                   // count = 1 valid descriptor
    g0.y = lds_addr;                             // LDS byte address
    g0.z = (unsigned)ga;                         // global addr [31:0]
    g0.w = (unsigned)((ga >> 32) & 0x1FFFFFFu)   // global addr [56:32]
         | (2u << 30);                           // type = 2
    return g0;
}

// issue a 2D tensor_load_to_lds (groups 2/3 unused -> zeros)
#if __has_include(<hip/amd_detail/amd_gfx1250_TDM.h>)
// therock / clang-23 lane: 6-arg builtin
#define TDM_LOAD(g0, g1)                                               \
    do {                                                               \
        tdm4i z4 = {0, 0, 0, 0};                                       \
        tdm8i z8 = {0, 0, 0, 0, 0, 0, 0, 0};                           \
        __builtin_amdgcn_tensor_load_to_lds((g0), (g1), z4, z4, z8, 0);\
    } while (0)
#else
// ROCm 7.2 / clang-22 lane: 5-arg builtin
#define TDM_LOAD(g0, g1)                                               \
    do {                                                               \
        tdm4i z4 = {0, 0, 0, 0};                                       \
        __builtin_amdgcn_tensor_load_to_lds((g0), (g1), z4, z4, 0);    \
    } while (0)
#endif

// D# group1 word0: data_size=2B (code 1), pad_enable, pad_interval=3
// (pad after 16 DWORDs = 64B row), pad_amount=0 (+1 DWORD) ->
// reproduces the [rows][34 bf16] padded LDS layout in hardware.
#define TDM_G1_W0  0x00D10000

// =====================================================================
// Kernel 1: LayerNorm over feature dim + cast to bf16
// =====================================================================
__global__ __launch_bounds__(256)
void ln_cast_kernel(const float* __restrict__ x,
                    const float* __restrict__ gamma,
                    const float* __restrict__ beta,
                    __bf16* __restrict__ q)
{
    __shared__ float red[256];
    const int row = blockIdx.x;
    const int t   = threadIdx.x;
    const float* xr = x + (size_t)row * INFEA;

    float x0 = xr[t];
    float x1 = xr[t + 256];

    red[t] = x0 + x1;
    __syncthreads();
    for (int s = 128; s > 0; s >>= 1) {
        if (t < s) red[t] += red[t + s];
        __syncthreads();
    }
    const float mean = red[0] * (1.0f / INFEA);
    __syncthreads();

    const float d0 = x0 - mean, d1 = x1 - mean;
    red[t] = d0 * d0 + d1 * d1;
    __syncthreads();
    for (int s = 128; s > 0; s >>= 1) {
        if (t < s) red[t] += red[t + s];
        __syncthreads();
    }
    const float var = red[0] * (1.0f / INFEA);
    const float rs  = rsqrtf(var + LN_EPS);

    q[(size_t)row * INFEA + t]       = (__bf16)(d0 * rs * gamma[t]       + beta[t]);
    q[(size_t)row * INFEA + t + 256] = (__bf16)(d1 * rs * gamma[t + 256] + beta[t + 256]);
}

// =====================================================================
// Kernel 2: cast per-head projection weights W to bf16
// =====================================================================
__global__ __launch_bounds__(256)
void cast_w_kernel(const float* __restrict__ W, __bf16* __restrict__ wb, int n)
{
    int i = blockIdx.x * 256 + threadIdx.x;
    if (i < n) wb[i] = (__bf16)W[i];
}

// =====================================================================
// Kernel 3: projection GEMM   h[head, m, o] = q[m, :] @ W[head, :, o]
// A tile (128x32 bf16) staged by the Tensor Data Mover with HW padding;
// B tile staged cooperatively (transposed). 8 waves x (16 rows x 128 cols).
// =====================================================================
__global__ __launch_bounds__(256)
void proj_gemm_kernel(const __bf16* __restrict__ qb,
                      const __bf16* __restrict__ wb,
                      float*  __restrict__ hf,
                      __bf16* __restrict__ hT)
{
    const int head = blockIdx.y;
    const int m0   = blockIdx.x * 128;

    __shared__ __bf16 a_sm[128][34];   // q tile [row][k], stride padded by TDM
    __shared__ __bf16 b_sm[128][34];   // W tile transposed [col][k], padded

    const int t    = threadIdx.x;
    const int wave = t >> 5;
    const int lane = t & 31;
    const int half = lane >> 4;
    const int mr   = lane & 15;

    // TDM descriptor group 1 (uniform): tensor q = [4096][512] bf16,
    // tile 32 (k) x 128 (rows), stride0 = 512
    tdm8i g1;
    g1[0] = TDM_G1_W0;
    g1[1] = (int)(512u  << 16);        // tensor_dim0 = 512
    g1[2] = (int)(4096u << 16);        // tensor_dim1 = 4096
    g1[3] = (int)(32u   << 16);        // tile_dim0   = 32
    g1[4] = 128;                       // tile_dim1   = 128
    g1[5] = 512;                       // tensor_dim0_stride = 512
    g1[6] = 0;
    g1[7] = 0;
    const unsigned a_lds = (unsigned)(uintptr_t)&a_sm[0][0];

    v8f acc[8];
    #pragma unroll
    for (int i = 0; i < 8; ++i) acc[i] = v8f{0,0,0,0,0,0,0,0};

    for (int k0 = 0; k0 < INFEA; k0 += 32) {
        // ---- stage A via TDM (wave 0 issues one DMA) ----
        if (wave == 0) {
            tdm4u g0 = tdm_make_g0(a_lds, qb + (size_t)m0 * INFEA + k0);
            TDM_LOAD(g0, g1);
        }
        // ---- stage B (transposed) cooperatively ----
        {
            const int kr = t >> 3;            // 0..31
            const int cs = (t & 7) * 16;      // 0..112
            const __bf16* g = wb + ((size_t)head * INFEA + k0 + kr) * HIDDEN + cs;
            #pragma unroll
            for (int c = 0; c < 16; ++c) b_sm[cs + c][kr] = g[c];
        }
        if (wave == 0) __builtin_amdgcn_s_wait_tensorcnt(0);
        __syncthreads();

        // ---- A fragment: rows wave*16 .. wave*16+15, K = k0..k0+31 ----
        Frag16 af;
        const int arow = wave * 16 + mr;
        #pragma unroll
        for (int v = 0; v < 8; ++v) {
            const int kb = ((v < 4) ? (2 * v) : (16 + 2 * (v - 4))) + half * 8;
            af.u[v] = *(const unsigned int*)&a_sm[arow][kb];
        }
        // ---- 8 N-tiles of 16 columns ----
        #pragma unroll
        for (int nt = 0; nt < 8; ++nt) {
            Frag16 bf;
            const int col = nt * 16 + mr;
            #pragma unroll
            for (int v = 0; v < 8; ++v) {
                const int kk = ((v < 4) ? (2 * v) : (16 + 2 * (v - 4))) + half * 8;
                bf.u[v] = *(const unsigned int*)&b_sm[col][kk];
            }
            acc[nt] = __builtin_amdgcn_wmma_f32_16x16x32_bf16(
                false, af.v, false, bf.v, (short)0, acc[nt], false, false);
        }
        __syncthreads();
    }

    // ---- epilogue: write h (f32) and transposed bf16 copy ----
    #pragma unroll
    for (int nt = 0; nt < 8; ++nt) {
        #pragma unroll
        for (int r = 0; r < 8; ++r) {
            const int m = m0 + wave * 16 + r + half * 8;
            const int o = nt * 16 + mr;
            const float val = acc[nt][r];
            hf[((size_t)head * Mrows + m) * HIDDEN + o] = val;
            const int b = m >> 10, n = m & 1023;
            hT[(((size_t)(head * Bdim + b) * HIDDEN) + o) * Ndim + n] = (__bf16)val;
        }
    }
}

// =====================================================================
// Kernel 4: s_src/s_dst  [NHEAD][NEDGE][Bdim][Ndim]
// =====================================================================
__global__ __launch_bounds__(256)
void s_proj_kernel(const float* __restrict__ hf,
                   const float* __restrict__ wsrc,
                   const float* __restrict__ wdst,
                   float* __restrict__ ssrc,
                   float* __restrict__ sdst)
{
    const int idx  = blockIdx.x * 256 + threadIdx.x;   // 0..16383
    const int head = idx >> 12;
    const int m    = idx & (Mrows - 1);

    float as[NEDGE], ad[NEDGE];
    #pragma unroll
    for (int e = 0; e < NEDGE; ++e) { as[e] = 0.f; ad[e] = 0.f; }

    const float* hr = hf   + ((size_t)head * Mrows + m) * HIDDEN;
    const float* ws = wsrc + (size_t)head * NEDGE * HIDDEN;
    const float* wd = wdst + (size_t)head * NEDGE * HIDDEN;

    for (int o = 0; o < HIDDEN; ++o) {
        const float hv = hr[o];
        #pragma unroll
        for (int e = 0; e < NEDGE; ++e) {
            as[e] = fmaf(hv, ws[e * HIDDEN + o], as[e]);
            ad[e] = fmaf(hv, wd[e * HIDDEN + o], ad[e]);
        }
    }
    const int b = m >> 10, n = m & 1023;
    #pragma unroll
    for (int e = 0; e < NEDGE; ++e) {
        ssrc[((size_t)(head * NEDGE + e) * Bdim + b) * Ndim + n] = as[e];
        sdst[((size_t)(head * NEDGE + e) * Bdim + b) * Ndim + n] = ad[e];
    }
}

// =====================================================================
// Kernel 5: fused edge-score -> softmax -> (P @ h) -> relu
// P (unnormalized exp, bf16) built in LDS; PV GEMM consumes h tiles
// double-buffered into LDS by the Tensor Data Mover (TENSORcnt pipelining).
// =====================================================================
__global__ __launch_bounds__(256)
void attn_kernel(const int*   __restrict__ adj,
                 const float* __restrict__ ssrc,
                 const float* __restrict__ sdst,
                 const float* __restrict__ bedge,
                 const __bf16* __restrict__ hT,
                 float* __restrict__ out)
{
    const int i0 = blockIdx.x * 16;
    const int b  = blockIdx.y;
    const int h  = blockIdx.z;

    __shared__ __bf16 p_sm[16][1032];       // unnormalized probs, padded stride
    __shared__ __bf16 b_tile[2][128][34];   // double-buffered h chunks (TDM)
    __shared__ float  rowsum[16];
    __shared__ float  ssrc_sm[NEDGE][16];
    __shared__ float  bed_sm[NEDGE];

    const int t = threadIdx.x;
    if (t < NEDGE * 16) {
        const int e = t >> 4, il0 = t & 15;
        ssrc_sm[e][il0] = ssrc[((size_t)(h * NEDGE + e) * Bdim + b) * Ndim + i0 + il0];
    }
    if (t >= 80 && t < 80 + NEDGE) bed_sm[t - 80] = bedge[h * NEDGE + (t - 80)];
    __syncthreads();

    // ---- score passes: 16 threads per row, 64 j's each ----
    const int il = t >> 4;                  // local row 0..15
    const int jl = t & 15;
    const int*   adjrow = adj  + ((size_t)b * Ndim + i0 + il) * Ndim;
    const float* sdr    = sdst + (size_t)(h * NEDGE * Bdim + b) * Ndim;

    // pass 1: row max
    float mmax = -3.0e38f;
    for (int jj = 0; jj < 64; ++jj) {
        const int j = jj * 16 + jl;
        const int a = adjrow[j];
        float w;
        if (a == 0) w = NEG_INF;
        else {
            const int e = a - 1;
            w = ssrc_sm[e][il] + sdr[(size_t)e * (Bdim * Ndim) + j] + bed_sm[e];
            w = (w >= 0.f) ? w : SLOPE * w;
        }
        mmax = fmaxf(mmax, w);
    }
    #pragma unroll
    for (int off = 1; off < 16; off <<= 1)
        mmax = fmaxf(mmax, __shfl_xor(mmax, off, 32));

    // pass 2: exp, accumulate sum, store bf16 P
    float ssum = 0.f;
    for (int jj = 0; jj < 64; ++jj) {
        const int j = jj * 16 + jl;
        const int a = adjrow[j];
        float w;
        if (a == 0) w = NEG_INF;
        else {
            const int e = a - 1;
            w = ssrc_sm[e][il] + sdr[(size_t)e * (Bdim * Ndim) + j] + bed_sm[e];
            w = (w >= 0.f) ? w : SLOPE * w;
        }
        const float p = __expf(w - mmax);
        ssum += p;
        p_sm[il][j] = (__bf16)p;
    }
    #pragma unroll
    for (int off = 1; off < 16; off <<= 1)
        ssum += __shfl_xor(ssum, off, 32);
    if (jl == 0) rowsum[il] = ssum;
    __syncthreads();

    // ---- PV GEMM: wave w owns output cols [w*16, w*16+16) ----
    const int wave = t >> 5;
    const int lane = t & 31;
    const int half = lane >> 4;
    const int mr   = lane & 15;
    const int o0   = wave * 16;

    // TDM descriptor group 1 (uniform): tensor hT slab = [128][1024] bf16,
    // tile 32 (k) x 128 (o rows), stride0 = 1024
    tdm8i g1;
    g1[0] = TDM_G1_W0;
    g1[1] = (int)(1024u << 16);        // tensor_dim0 = 1024
    g1[2] = (int)(128u  << 16);        // tensor_dim1 = 128
    g1[3] = (int)(32u   << 16);        // tile_dim0   = 32
    g1[4] = 128;                       // tile_dim1   = 128
    g1[5] = 1024;                      // tensor_dim0_stride = 1024
    g1[6] = 0;
    g1[7] = 0;
    const __bf16* slab = hT + ((size_t)(h * Bdim + b) * HIDDEN) * Ndim;
    const unsigned bt_lds0 = (unsigned)(uintptr_t)&b_tile[0][0][0];
    const unsigned bt_lds1 = (unsigned)(uintptr_t)&b_tile[1][0][0];

    if (wave == 0) {                   // prime the pipeline: chunk 0
        tdm4u g0 = tdm_make_g0(bt_lds0, slab);
        TDM_LOAD(g0, g1);
    }

    v8f acc = v8f{0,0,0,0,0,0,0,0};
    for (int kk = 0; kk < 32; ++kk) {
        if (wave == 0) {
            if (kk + 1 < 32) {         // issue next chunk, wait for current
                tdm4u g0 = tdm_make_g0(((kk + 1) & 1) ? bt_lds1 : bt_lds0,
                                       slab + (kk + 1) * 32);
                TDM_LOAD(g0, g1);
                __builtin_amdgcn_s_wait_tensorcnt(1);
            } else {
                __builtin_amdgcn_s_wait_tensorcnt(0);
            }
        }
        __syncthreads();

        const int kO = kk * 32;
        Frag16 av, bv;
        #pragma unroll
        for (int v = 0; v < 8; ++v) {
            const int kb = ((v < 4) ? (2 * v) : (16 + 2 * (v - 4))) + half * 8;
            av.u[v] = *(const unsigned int*)&p_sm[mr][kO + kb];
            const int kl = ((v < 4) ? (2 * v) : (16 + 2 * (v - 4))) + half * 8;
            bv.u[v] = *(const unsigned int*)&b_tile[kk & 1][o0 + mr][kl];
        }
        acc = __builtin_amdgcn_wmma_f32_16x16x32_bf16(
            false, av.v, false, bv.v, (short)0, acc, false, false);
        __syncthreads();
    }

    // ---- epilogue: normalize, relu, scatter to [B, N, H*HID] ----
    #pragma unroll
    for (int r = 0; r < 8; ++r) {
        const int ir  = r + half * 8;
        float val = acc[r] / rowsum[ir];
        val = fmaxf(val, 0.0f);
        out[((size_t)b * Ndim + i0 + ir) * (NHEAD * HIDDEN) + h * HIDDEN + o0 + mr] = val;
    }
}

// =====================================================================
// launcher
// =====================================================================
extern "C" void kernel_launch(void* const* d_in, const int* in_sizes, int n_in,
                              void* d_out, int out_size, void* d_ws, size_t ws_size,
                              hipStream_t stream)
{
    const float* nodes = (const float*)d_in[0];
    const int*   adj   = (const int*)  d_in[1];
    // d_in[2] node_info, d_in[3] global_step: unused by the reference math
    const float* W     = (const float*)d_in[4];
    const float* wsrc  = (const float*)d_in[5];
    const float* wdst  = (const float*)d_in[6];
    const float* bedge = (const float*)d_in[7];
    const float* gamma = (const float*)d_in[8];
    const float* beta  = (const float*)d_in[9];
    float* out = (float*)d_out;

    char* ws = (char*)d_ws;
    size_t off = 0;
    __bf16* qb   = (__bf16*)(ws + off); off += (size_t)Mrows * INFEA * 2;                 // 4 MB
    __bf16* wb   = (__bf16*)(ws + off); off += (size_t)NHEAD * INFEA * HIDDEN * 2;        // 0.5 MB
    float*  hf   = (float*) (ws + off); off += (size_t)NHEAD * Mrows * HIDDEN * 4;        // 8 MB
    __bf16* hT   = (__bf16*)(ws + off); off += (size_t)NHEAD * Bdim * HIDDEN * Ndim * 2;  // 4 MB
    float*  ssrc = (float*) (ws + off); off += (size_t)NHEAD * NEDGE * Bdim * Ndim * 4;
    float*  sdst = (float*) (ws + off); off += (size_t)NHEAD * NEDGE * Bdim * Ndim * 4;
    (void)ws_size; (void)n_in; (void)in_sizes; (void)out_size;

    // 1. LayerNorm + bf16 cast
    ln_cast_kernel<<<Mrows, 256, 0, stream>>>(nodes, gamma, beta, qb);

    // 2. cast W
    const int nW = NHEAD * INFEA * HIDDEN;
    cast_w_kernel<<<(nW + 255) / 256, 256, 0, stream>>>(W, wb, nW);

    // 3. projection GEMM (bf16 WMMA + TDM-staged A)
    proj_gemm_kernel<<<dim3(Mrows / 128, NHEAD), 256, 0, stream>>>(qb, wb, hf, hT);

    // 4. edge score projections
    s_proj_kernel<<<(NHEAD * Mrows) / 256, 256, 0, stream>>>(hf, wsrc, wdst, ssrc, sdst);

    // 5. fused score/softmax/PV (bf16 WMMA + double-buffered TDM)
    attn_kernel<<<dim3(Ndim / 16, Bdim, NHEAD), 256, 0, stream>>>(
        adj, ssrc, sdst, bedge, hT, out);
}
